// GNN4GAE_35261681500245
// MI455X (gfx1250) — matrix operand order repaired and checked by reading, not compiled, not run on hardware
//
#include <hip/hip_runtime.h>
#include <hip/hip_bf16.h>
#include <math.h>

// Fixed hidden width for this workload (HID == 128 == 8 WMMA column-tiles,
// and 32 lanes x float4 per node row).
#define HIDW 128

typedef __attribute__((ext_vector_type(16))) _Float16 v16h;
typedef __attribute__((ext_vector_type(8)))  _Float16 v8h;
typedef __attribute__((ext_vector_type(4)))  _Float16 v4h;
typedef __attribute__((ext_vector_type(8)))  float    v8f;

// ---------------------------------------------------------------------------
// Order-preserving float <-> uint encoding for atomicMax on floats.
__device__ __forceinline__ unsigned fenc(float f) {
    unsigned u = __float_as_uint(f);
    return (u & 0x80000000u) ? ~u : (u | 0x80000000u);
}
__device__ __forceinline__ float fdec(unsigned e) {
    unsigned u = (e & 0x80000000u) ? (e ^ 0x80000000u) : ~e;
    return __uint_as_float(u);
}
#define ENC_NEG_INF 0x007FFFFFu  // fenc(-inf)

// ---------------------------------------------------------------------------
// One-time prepack: Wh[n][k] = (f16) W[k][n].   W: [K,128] f32 row-major.
// Tiny (<= 32K elements); keeps the GEMM free of LDS transposes.
__global__ __launch_bounds__(256)
void pack_w_f16(const float* __restrict__ W, _Float16* __restrict__ Wh, int K)
{
    int idx = blockIdx.x * 256 + threadIdx.x;
    if (idx >= K * HIDW) return;
    int n = idx / K;
    int k = idx - n * K;
    Wh[idx] = (_Float16)W[(size_t)k * HIDW + n];
}

// ---------------------------------------------------------------------------
// C[M,128] = A[M,K] (f32) @ W[K,128] (+ bias); f16 WMMA, f32 accumulate.
// Block = 256 threads = 8 waves. Block owns a 16-row tile; wave w owns
// columns [16w,16w+16). The whole 16xK A tile is staged in LDS as f16 ONCE
// (float4 loads -> packed b64 LDS stores, fully unrolled), then the K loop
// is barrier-free: 2x ds_load_b128 (A frag) + 2x global b128 (B frag from
// the prepacked column-major f16 Wh, L2/L0-resident) + one v_wmma per chunk.
template <int K>
__global__ __launch_bounds__(256)
void gemm128_wmma(const float* __restrict__ A, const _Float16* __restrict__ Wh,
                  const float* __restrict__ bias, float* __restrict__ C, int M)
{
    constexpr int KP = K + 8;              // pad: row stride 528/272 B (16B-aligned)
    __shared__ _Float16 As[16][KP];

    const int m0   = blockIdx.x * 16;
    const int lane = threadIdx.x & 31;
    const int wave = threadIdx.x >> 5;
    const int n0   = wave * 16;

    // ---- Stage A tile (16 x K) as f16; float4 -> v4h (ds_store_b64). ----
    constexpr int QPR = K / 4;             // float4 quads per row
#pragma unroll
    for (int i = threadIdx.x; i < 16 * QPR; i += 256) {
        int r = i / QPR;                   // QPR is a power of two -> shifts
        int k = (i - r * QPR) * 4;
        float4 v = {0.0f, 0.0f, 0.0f, 0.0f};
        if (m0 + r < M) v = *(const float4*)(A + (size_t)(m0 + r) * K + k);
        v4h p = {(_Float16)v.x, (_Float16)v.y, (_Float16)v.z, (_Float16)v.w};
        *(v4h*)&As[r][k] = p;
    }
    __syncthreads();

    // ---- Fragment coordinates (ISA 7.12.2 layouts, wave32). ----
    const int ar = lane & 15;              // A: row
    const int kA = (lane >> 4) * 8;        // A: elems 0-7 -> K=kA.., 8-15 -> K=16+kA..
    const int bc = lane & 15;              // B: column
    const int kB = (lane >> 4) * 8;        // B: elems 0-7 -> K=kB.., 8-15 -> K=16+kB..
    const _Float16* wrow = Wh + (size_t)(n0 + bc) * K;

    v8f acc = {};
#pragma unroll
    for (int kc = 0; kc < K; kc += 32) {
        v8h alo = *(const v8h*)&As[ar][kc + kA];
        v8h ahi = *(const v8h*)&As[ar][kc + 16 + kA];
        v16h afrag = __builtin_shufflevector(alo, ahi,
            0,1,2,3,4,5,6,7,8,9,10,11,12,13,14,15);

        v8h blo = *(const v8h*)(wrow + kc + kB);
        v8h bhi = *(const v8h*)(wrow + kc + 16 + kB);
        v16h bfrag = __builtin_shufflevector(blo, bhi,
            0,1,2,3,4,5,6,7,8,9,10,11,12,13,14,15);

        acc = __builtin_amdgcn_wmma_f32_16x16x32_f16(
            false, afrag, false, bfrag, (short)0, acc, false, false);
    }

    // ---- D layout: col = lane&15, row = (lane>>4)*8 + r. ----
    const int col   = n0 + (lane & 15);
    const int rbase = (lane >> 4) * 8;
    const float bv  = bias ? bias[col] : 0.0f;
#pragma unroll
    for (int r = 0; r < 8; ++r) {
        int gm = m0 + rbase + r;
        if (gm < M) C[(size_t)gm * HIDW + col] = acc[r] + bv;
    }
}

// ---------------------------------------------------------------------------
// Per-node attention scores: ssrc[i] = hp[i].a_src ; sdst[i] = hp[i].a_dst.
// One wave per node; lane handles 4 features; shuffle reduction (wave32).
__global__ __launch_bounds__(256)
void node_scores(const float* __restrict__ hp,
                 const float* __restrict__ asrc, const float* __restrict__ adst,
                 float* __restrict__ ssrc, float* __restrict__ sdst, int N)
{
    int node = blockIdx.x * 8 + (threadIdx.x >> 5);
    if (node >= N) return;
    int lane = threadIdx.x & 31;
    const float4 h4 = *(const float4*)(hp + (size_t)node * HIDW + lane * 4);
    const float4 a4 = *(const float4*)(asrc + lane * 4);
    const float4 b4 = *(const float4*)(adst + lane * 4);
    float s = h4.x * a4.x + h4.y * a4.y + h4.z * a4.z + h4.w * a4.w;
    float t = h4.x * b4.x + h4.y * b4.y + h4.z * b4.z + h4.w * b4.w;
#pragma unroll
    for (int off = 16; off > 0; off >>= 1) {
        s += __shfl_xor(s, off);
        t += __shfl_xor(t, off);
    }
    if (lane == 0) { ssrc[node] = s; sdst[node] = t; }
}

// ---------------------------------------------------------------------------
// Per-layer init: tmp = 0, segment-max = enc(-inf), denom = 0.
__global__ __launch_bounds__(256)
void layer_init(float* __restrict__ tmp, unsigned* __restrict__ menc,
                float* __restrict__ denom, int N)
{
    int idx = blockIdx.x * 256 + threadIdx.x;
    if (idx < N * HIDW) tmp[idx] = 0.0f;
    if (idx < N) { menc[idx] = ENC_NEG_INF; denom[idx] = 0.0f; }
}

// Pass 1: logit = leaky_relu(ssrc[src] + sdst[dst]); segment max over dst.
__global__ __launch_bounds__(256)
void edge_logit(const int* __restrict__ src, const int* __restrict__ dst,
                const float* __restrict__ ssrc, const float* __restrict__ sdst,
                float* __restrict__ logit, unsigned* __restrict__ menc, int E)
{
    int e = blockIdx.x * 256 + threadIdx.x;
    if (e >= E) return;
    float x  = ssrc[src[e]] + sdst[dst[e]];
    float lg = x > 0.0f ? x : 0.2f * x;
    logit[e] = lg;
    atomicMax(menc + dst[e], fenc(lg));
}

// Pass 2: e = exp(logit - m[dst]) * ew ; segment sum over dst. (in-place buf)
__global__ __launch_bounds__(256)
void edge_exp(const int* __restrict__ dst, const float* __restrict__ ew,
              float* __restrict__ ebuf, const unsigned* __restrict__ menc,
              float* __restrict__ denom, int E)
{
    int e = blockIdx.x * 256 + threadIdx.x;
    if (e >= E) return;
    int d    = dst[e];
    float ev = __expf(ebuf[e] - fdec(menc[d])) * ew[e];
    ebuf[e]  = ev;
    atomicAdd(denom + d, ev);
}

// Pass 3: tmp[dst] += hp[src] * (e / (denom[dst] + 1e-16)).  Wave per edge,
// lane covers 4 features; hp is L2-resident (25.6 MB << 192 MB L2).
__global__ __launch_bounds__(256)
void edge_aggregate(const int* __restrict__ src, const int* __restrict__ dst,
                    const float* __restrict__ ebuf, const float* __restrict__ denom,
                    const float* __restrict__ hp, float* __restrict__ tmp, int E)
{
    int e = blockIdx.x * 8 + (threadIdx.x >> 5);
    if (e >= E) return;
    int lane = threadIdx.x & 31;
    int s = src[e], d = dst[e];
    __builtin_prefetch(hp + (size_t)s * HIDW, 0, 1);
    float alpha = ebuf[e] / (denom[d] + 1e-16f);
    const float4 h4 = *(const float4*)(hp + (size_t)s * HIDW + lane * 4);
    float* o = tmp + (size_t)d * HIDW + lane * 4;
    atomicAdd(o + 0, h4.x * alpha);
    atomicAdd(o + 1, h4.y * alpha);
    atomicAdd(o + 2, h4.z * alpha);
    atomicAdd(o + 3, h4.w * alpha);
}

// ---------------------------------------------------------------------------
// Residual + running elementwise max into out.
// mode 0: h += tmp+b; out = h           (first of several layers)
// mode 1: h += tmp+b; out = max(out,h)  (middle layer)
// mode 2: out = max(out, tmp+b)         (last layer, no residual)
// mode 3: out = tmp+b                   (single-layer edge case)
__global__ __launch_bounds__(256)
void finalize(float* __restrict__ h, const float* __restrict__ tmp,
              const float* __restrict__ bias, float* __restrict__ out,
              int N, int numNode, int mode)
{
    int idx = blockIdx.x * 256 + threadIdx.x;
    if (idx >= N * HIDW) return;
    int node = idx >> 7;
    int f    = idx & (HIDW - 1);
    float t  = tmp[idx] + bias[f];
    if (mode <= 1) {
        float nh = h[idx] + t;
        h[idx] = nh;
        if (node < numNode)
            out[idx] = (mode == 0) ? nh : fmaxf(out[idx], nh);
    } else {
        if (node < numNode)
            out[idx] = (mode == 3) ? t : fmaxf(out[idx], t);
    }
}

// ---------------------------------------------------------------------------
static inline void launch_gemm(const float* A, const _Float16* Wh,
                               const float* bias, float* C, int M, int K,
                               hipStream_t stream)
{
    int grid = (M + 15) / 16;
    if (K == 256)
        gemm128_wmma<256><<<grid, 256, 0, stream>>>(A, Wh, bias, C, M);
    else
        gemm128_wmma<128><<<grid, 256, 0, stream>>>(A, Wh, bias, C, M);
}

extern "C" void kernel_launch(void* const* d_in, const int* in_sizes, int n_in,
                              void* d_out, int out_size, void* d_ws, size_t ws_size,
                              hipStream_t stream)
{
    (void)n_in; (void)ws_size;
    const float* x       = (const float*)d_in[0];
    const int*   ei      = (const int*)d_in[1];
    const float* ew      = (const float*)d_in[2];
    // d_in[3] = numNode (device scalar; recovered from out_size instead)
    const float* Wlin    = (const float*)d_in[4];
    const float* blin    = (const float*)d_in[5];
    const float* Wc      = (const float*)d_in[6];
    const float* att_src = (const float*)d_in[7];
    const float* att_dst = (const float*)d_in[8];
    const float* bias_c  = (const float*)d_in[9];
    float* out = (float*)d_out;

    const int HID     = in_sizes[5];                 // 128
    const int IN_FT   = in_sizes[4] / HID;           // 256
    const int N       = in_sizes[0] / IN_FT;         // 50000
    const int E       = in_sizes[2];                 // 800000
    const int L       = in_sizes[6] / (HID * HID);   // 3
    const int numNode = out_size / HID;
    const int* src = ei;
    const int* dst = ei + E;

    // Workspace carve-up (~81 MB total).
    char* ws = (char*)d_ws;
    size_t off = 0;
    auto carve = [&](size_t bytes) -> void* {
        void* p = ws + off;
        off = (off + bytes + 255) & ~(size_t)255;
        return p;
    };
    float*     h     = (float*)carve((size_t)N * HID * 4);
    float*     hp    = (float*)carve((size_t)N * HID * 4);
    float*     tmp   = (float*)carve((size_t)N * HID * 4);
    float*     ssrc  = (float*)carve((size_t)N * 4);
    float*     sdst  = (float*)carve((size_t)N * 4);
    unsigned*  menc  = (unsigned*)carve((size_t)N * 4);
    float*     denom = (float*)carve((size_t)N * 4);
    float*     ebuf  = (float*)carve((size_t)E * 4);
    _Float16*  Wh    = (_Float16*)carve((size_t)HIDW * 256 * 2);  // packed f16 W

    const int gNF = (N * HID + 255) / 256;
    const int gE  = (E + 255) / 256;
    const int gEw = (E + 7) / 8;
    const int gNw = (N + 7) / 8;

    // h = x @ Wlin + blin
    pack_w_f16<<<(IN_FT * HIDW + 255) / 256, 256, 0, stream>>>(Wlin, Wh, IN_FT);
    launch_gemm(x, Wh, blin, h, N, IN_FT, stream);

    for (int l = 0; l < L; ++l) {
        const float* W  = Wc + (size_t)l * HID * HID;
        const float* as = att_src + (size_t)l * HID;
        const float* ad = att_dst + (size_t)l * HID;
        const float* bc = bias_c + (size_t)l * HID;

        pack_w_f16<<<(HID * HIDW + 255) / 256, 256, 0, stream>>>(W, Wh, HID);
        launch_gemm(h, Wh, nullptr, hp, N, HID, stream);

        node_scores<<<gNw, 256, 0, stream>>>(hp, as, ad, ssrc, sdst, N);
        layer_init<<<gNF, 256, 0, stream>>>(tmp, menc, denom, N);
        edge_logit<<<gE, 256, 0, stream>>>(src, dst, ssrc, sdst, ebuf, menc, E);
        edge_exp<<<gE, 256, 0, stream>>>(dst, ew, ebuf, menc, denom, E);
        edge_aggregate<<<gEw, 256, 0, stream>>>(src, dst, ebuf, denom, hp, tmp, E);

        int mode;
        if (l == L - 1) mode = (L == 1) ? 3 : 2;
        else            mode = (l == 0) ? 0 : 1;
        finalize<<<gNF, 256, 0, stream>>>(h, tmp, bc, out, N, numNode, mode);
    }
}
